// Network_89421219103170
// MI455X (gfx1250) — matrix operand order repaired and checked by reading, not compiled, not run on hardware
//
#include <hip/hip_runtime.h>
#include <hip/hip_bf16.h>

typedef __attribute__((ext_vector_type(16))) _Float16 v16h;
typedef __attribute__((ext_vector_type(8)))  _Float16 v8h;
typedef __attribute__((ext_vector_type(8)))  float    v8f;

#define T_STEPS 512
#define ROWS    16384          // B*N = 256*64
#define NNEUR   64
#define WPB     4              // waves per block
#define WHH_PAD 72             // halves per W_hh row (padded, bank-friendly)
#define H_PAD   72             // halves per h row (padded)

__device__ __forceinline__ float sigm_f(float x) {
    return __builtin_amdgcn_rcpf(1.0f + __expf(-x));
}
__device__ __forceinline__ float tanh_f(float x) {
    // tanh(x) = 2*sigmoid(2x) - 1
    return 2.0f * sigm_f(2.0f * x) - 1.0f;
}

__global__ __launch_bounds__(128)
void lstm_wmma_kernel(const float* __restrict__ input,
                      const float* __restrict__ w1,
                      const float* __restrict__ b1,
                      const float* __restrict__ w2,
                      const float* __restrict__ b2,
                      const float* __restrict__ W_ih,
                      const float* __restrict__ W_hh,
                      const float* __restrict__ b_ih,
                      const float* __restrict__ b_hh,
                      const float* __restrict__ wg,
                      const float* __restrict__ bg,
                      float* __restrict__ out)
{
    __shared__ _Float16 whh_lds[256 * WHH_PAD];        // 36864 B
    __shared__ _Float16 h_lds[WPB * 16 * H_PAD];       //  9216 B
    __shared__ float    g_lds[WPB * 16];               //   256 B

    const int tid = threadIdx.x;

    // Stage W_hh (f32 [256][64]) -> LDS f16 padded row-major.
    for (int i = tid; i < 256 * 64; i += 128) {
        int g = i >> 6, k = i & 63;
        whh_lds[g * WHH_PAD + k] = (_Float16)W_hh[i];
    }
    // Zero initial hidden state.
    for (int i = tid; i < WPB * 16 * H_PAD; i += 128) h_lds[i] = (_Float16)0.0f;
    __syncthreads();

    const int wave = tid >> 5;
    const int lane = tid & 31;
    const int lid  = lane & 15;     // within-16 lane id
    const int hi   = lane >> 4;     // 0 or 1 (half-wave)
    const int mrow = hi << 3;       // D-layout row base (0 or 8)

    const int r0  = (blockIdx.x * WPB + wave) * 16;    // first global row of this wave
    const int myn = (r0 & (NNEUR - 1)) + lid;          // neuron index for this lane's row

    // Per-lane per-neuron MLP params (Linear(1,10) -> Linear(10,1)).
    float w1r[10], b1r[10], w2r[10];
    #pragma unroll
    for (int u = 0; u < 10; ++u) {
        w1r[u] = w1[myn * 10 + u];
        b1r[u] = b1[myn * 10 + u];
        w2r[u] = w2[myn * 10 + u];
    }
    const float b2r = b2[myn];

    // Gate constants: lane lid covers gate column q*64 + cg*16 + lid.
    float wihr[4][4], biasr[4][4];
    #pragma unroll
    for (int q = 0; q < 4; ++q)
      #pragma unroll
      for (int cg = 0; cg < 4; ++cg) {
        int gi = q * 64 + cg * 16 + lid;
        wihr[q][cg]  = W_ih[gi];
        biasr[q][cg] = b_ih[gi] + b_hh[gi];
      }
    float wgl[4];
    #pragma unroll
    for (int cg = 0; cg < 4; ++cg) wgl[cg] = wg[cg * 16 + lid];
    const float bgs = bg[0];

    // Cell state in registers, D layout: cst[cg][v] = c[row mrow+v][hidden cg*16+lid].
    float cst[4][8];
    #pragma unroll
    for (int cg = 0; cg < 4; ++cg)
      #pragma unroll
      for (int v = 0; v < 8; ++v) cst[cg][v] = 0.0f;

    float gm1 = 0.0f, gm2 = 0.0f;        // g_{t-1}, g_{t-2} for row lid
    const int hbase = wave * 16 * H_PAD;
    const int gbase = wave * 16;

    #pragma unroll 1
    for (int t = 0; t < T_STEPS; ++t) {
        // ---- per-neuron MLP: x -> outLin (lane handles row r0+lid) ----
        float xin = input[(size_t)t * ROWS + r0 + lid];
        float xlin = b2r;
        #pragma unroll
        for (int u = 0; u < 10; ++u)
            xlin += (xin * w1r[u] + b1r[u]) * w2r[u];

        // Broadcast outLin to D-layout rows: xb[v] = outLin of row mrow+v.
        float xb[8];
        #pragma unroll
        for (int v = 0; v < 8; ++v) xb[v] = __shfl(xlin, mrow + v, 32);

        __syncthreads();   // previous step's h stores visible to all lanes

        // ---- load A fragments: h (16x64 f16), K split 0-31 / 32-63 ----
        const _Float16* hrow = &h_lds[hbase + lid * H_PAD];
        v8h a00 = *(const v8h*)(hrow +  0 + hi * 8);
        v8h a01 = *(const v8h*)(hrow + 16 + hi * 8);
        v8h a10 = *(const v8h*)(hrow + 32 + hi * 8);
        v8h a11 = *(const v8h*)(hrow + 48 + hi * 8);
        v16h afr[2];
        afr[0] = __builtin_shufflevector(a00, a01, 0,1,2,3,4,5,6,7,8,9,10,11,12,13,14,15);
        afr[1] = __builtin_shufflevector(a10, a11, 0,1,2,3,4,5,6,7,8,9,10,11,12,13,14,15);

        float gsum[8];
        #pragma unroll
        for (int v = 0; v < 8; ++v) gsum[v] = 0.0f;

        #pragma unroll
        for (int cg = 0; cg < 4; ++cg) {
            // Init accumulators with x*W_ih + (b_ih + b_hh), in D layout.
            v8f accq[4];
            #pragma unroll
            for (int q = 0; q < 4; ++q)
              #pragma unroll
              for (int v = 0; v < 8; ++v)
                accq[q][v] = xb[v] * wihr[q][cg] + biasr[q][cg];

            // gates[rows, q*64+cg*16+*] += h @ W_hh^T   (2 K-steps x 4 gates)
            #pragma unroll
            for (int kf = 0; kf < 2; ++kf) {
                #pragma unroll
                for (int q = 0; q < 4; ++q) {
                    const _Float16* wrow =
                        &whh_lds[(q * 64 + cg * 16 + lid) * WHH_PAD + kf * 32 + hi * 8];
                    v8h blo = *(const v8h*)(wrow);
                    v8h bhi = *(const v8h*)(wrow + 16);
                    v16h bf = __builtin_shufflevector(blo, bhi,
                              0,1,2,3,4,5,6,7,8,9,10,11,12,13,14,15);
                    accq[q] = __builtin_amdgcn_wmma_f32_16x16x32_f16(
                        false, afr[kf], false, bf, (short)0, accq[q], false, false);
                }
            }

            // Elementwise LSTM update for 16 hidden cols (cg group).
            #pragma unroll
            for (int v = 0; v < 8; ++v) {
                float ig = sigm_f(accq[0][v]);
                float fg = sigm_f(accq[1][v]);
                float gg = tanh_f(accq[2][v]);
                float og = sigm_f(accq[3][v]);
                float cc = fg * cst[cg][v] + ig * gg;
                cst[cg][v] = cc;
                float hn = og * tanh_f(cc);
                gsum[v] += hn * wgl[cg];
                h_lds[hbase + (mrow + v) * H_PAD + cg * 16 + lid] = (_Float16)hn;
            }
        }

        // ---- g_t = h . wg + bg : reduce across 16-lane groups, transpose via LDS ----
        #pragma unroll
        for (int v = 0; v < 8; ++v) {
            float s = gsum[v];
            s += __shfl_xor(s, 1, 32);
            s += __shfl_xor(s, 2, 32);
            s += __shfl_xor(s, 4, 32);
            s += __shfl_xor(s, 8, 32);
            if (lid == v) g_lds[gbase + mrow + v] = s;   // owner lanes v and 16+v
        }
        __syncthreads();
        float gcur = g_lds[gbase + lid] + bgs;

        // g_prev semantics: t=0 -> g_0 (current); t=1 -> 1; t>=2 -> g_{t-2}.
        float gprev = (t == 0) ? gcur : ((t == 1) ? 1.0f : gm2);
        gm2 = gm1; gm1 = gcur;

        float z  = xlin - 1.0f;
        float sp = (z > 20.0f) ? z : __logf(1.0f + __expf(z));
        if (lane < 16)
            out[(size_t)t * ROWS + r0 + lane] = gprev * sp;
    }
}

extern "C" void kernel_launch(void* const* d_in, const int* in_sizes, int n_in,
                              void* d_out, int out_size, void* d_ws, size_t ws_size,
                              hipStream_t stream) {
    (void)in_sizes; (void)n_in; (void)out_size; (void)d_ws; (void)ws_size;
    const float* input = (const float*)d_in[0];
    const float* w1    = (const float*)d_in[1];
    const float* b1    = (const float*)d_in[2];
    const float* w2    = (const float*)d_in[3];
    const float* b2    = (const float*)d_in[4];
    const float* W_ih  = (const float*)d_in[5];
    const float* W_hh  = (const float*)d_in[6];
    const float* b_ih  = (const float*)d_in[7];
    const float* b_hh  = (const float*)d_in[8];
    const float* wg    = (const float*)d_in[9];
    const float* bg    = (const float*)d_in[10];
    float* out = (float*)d_out;

    dim3 grid(ROWS / (WPB * 16));   // 256 blocks
    dim3 block(WPB * 32);           // 128 threads = 4 wave32
    lstm_wmma_kernel<<<grid, block, 0, stream>>>(
        input, w1, b1, w2, b2, W_ih, W_hh, b_ih, b_hh, wg, bg, out);
}
